// GCN_42451456753806
// MI455X (gfx1250) — matrix operand order repaired and checked by reading, not compiled, not run on hardware
//
#include <hip/hip_runtime.h>

#define N_NODES 10000
#define N_EDGES 160000
#define FEAT    1024          // HEADS * HID
#define HEADS   8
#define HID     128
#define NEG_INF_ORD ((int)0x807FFFFF)   // ordered-int encoding of -inf

typedef __attribute__((ext_vector_type(4)))  float        f32x4;
typedef __attribute__((ext_vector_type(8)))  float        f32x8;
typedef __attribute__((ext_vector_type(2)))  unsigned int u32x2;
typedef __attribute__((ext_vector_type(4)))  unsigned int u32x4;
typedef __attribute__((ext_vector_type(16))) __bf16       bf16x16;

union FragAB { u32x4 q[2]; bf16x16 v; };

// fp32 -> bf16 (round to nearest even)
__device__ inline unsigned short f2bf(float f) {
  unsigned int u = __float_as_uint(f);
  u += 0x7FFFu + ((u >> 16) & 1u);
  return (unsigned short)(u >> 16);
}
__device__ inline unsigned int pack2(float a, float b) {
  return (unsigned int)f2bf(a) | ((unsigned int)f2bf(b) << 16);
}
// monotone float<->int order mapping (for atomicMax on floats)
__device__ inline int ordF(float f) {
  int i = __float_as_int(f);
  return (i >= 0) ? i : (i ^ 0x7FFFFFFF);
}
__device__ inline float invOrd(int r) {
  int i = (r >= 0) ? r : (r ^ 0x7FFFFFFF);
  return __int_as_float(i);
}

// ---------------------------------------------------------------------------
// fp32 -> bf16 flat conversion (n multiple of 8)
// ---------------------------------------------------------------------------
__global__ __launch_bounds__(256) void cvt_bf16(
    const float* __restrict__ src, unsigned short* __restrict__ dst, int n)
{
  const int i = (blockIdx.x * 256 + threadIdx.x) * 8;
  if (i >= n) return;
  const f32x4 a = *(const f32x4*)(src + i);
  const f32x4 b = *(const f32x4*)(src + i + 4);
  u32x4 q = { pack2(a.x,a.y), pack2(a.z,a.w), pack2(b.x,b.y), pack2(b.z,b.w) };
  *(u32x4*)(dst + i) = q;
}

// ---------------------------------------------------------------------------
// W [K x 1024] fp32 row-major -> Wt [1024 x K] bf16 (transposed), 32x32 tiles
// ---------------------------------------------------------------------------
__global__ __launch_bounds__(256) void cvt_transpose_bf16(
    const float* __restrict__ W, unsigned short* __restrict__ Wt, int K)
{
  __shared__ unsigned short tile[32][36];   // 72B row pitch: 8B-aligned reads
  const int t  = threadIdx.x;
  const int tr = t >> 3;          // 0..31
  const int tc = (t & 7) << 2;    // 0..28 step 4
  const int k0 = blockIdx.x * 32;
  const int c0 = blockIdx.y * 32;
  const f32x4 v = *(const f32x4*)(W + (size_t)(k0 + tr) * FEAT + c0 + tc);
  tile[tc + 0][tr] = f2bf(v.x);
  tile[tc + 1][tr] = f2bf(v.y);
  tile[tc + 2][tr] = f2bf(v.z);
  tile[tc + 3][tr] = f2bf(v.w);
  __syncthreads();
  const unsigned int lo = (unsigned int)tile[tr][tc + 0] | ((unsigned int)tile[tr][tc + 1] << 16);
  const unsigned int hi = (unsigned int)tile[tr][tc + 2] | ((unsigned int)tile[tr][tc + 3] << 16);
  u32x2 q = { lo, hi };
  *(u32x2*)(Wt + (size_t)(c0 + tr) * K + k0 + tc) = q;
}

// ---------------------------------------------------------------------------
// GEMM: H[nRows x 1024] = A[nRows x K] @ B, bf16 WMMA, fp32 accum.
// A: bf16 row-major [nRows x K]; Bt: bf16 column-major [1024 x K] (W^T).
// Workgroup = 256 threads (8 waves) -> 128x64 output tile; K staged 32/step.
// ---------------------------------------------------------------------------
__global__ __launch_bounds__(256) void gat_gemm_bf16(
    const unsigned short* __restrict__ Abf, const unsigned short* __restrict__ Btbf,
    float* __restrict__ H, int nRows, int K)
{
  // 48-short (96B) row pitch: every row base 16B-aligned for b128 LDS ops
  __shared__ __align__(16) unsigned short As[128][48];
  __shared__ __align__(16) unsigned short Bs[64][48];

  const int t    = threadIdx.x;
  const int lane = t & 31;
  const int wave = t >> 5;
  const int rowBase = blockIdx.x * 128;
  const int colBase = blockIdx.y * 64;

  f32x8 acc0 = {0.f,0.f,0.f,0.f,0.f,0.f,0.f,0.f};
  f32x8 acc1 = acc0, acc2 = acc0, acc3 = acc0;

  // staging maps (all contiguous b128 copies)
  const int aRow = t >> 1;           // 0..127
  const int aOff = (t & 1) << 4;     // 0 or 16 shorts
  const int bCol = t >> 2;           // 0..63
  const int bOff = (t & 3) << 3;     // 0,8,16,24 shorts

  const int gRow = rowBase + aRow;
  const unsigned short* aSrc = Abf + (size_t)gRow * K + aOff;
  const unsigned short* bSrc = Btbf + (size_t)(colBase + bCol) * K + bOff;

  const int kSteps = K >> 5;
  for (int ks = 0; ks < kSteps; ++ks) {
    const int k0 = ks << 5;

    // ---- stage A tile: 128 x 32 bf16 ----
    {
      u32x4 q0 = {0u,0u,0u,0u}, q1 = q0;
      if (gRow < nRows) {
        const u32x4* p = (const u32x4*)(aSrc + k0);
        q0 = p[0]; q1 = p[1];
        if (k0 + 32 < K) __builtin_prefetch(aSrc + k0 + 32, 0, 3);  // next K-step
      }
      *(u32x4*)&As[aRow][aOff]     = q0;
      *(u32x4*)&As[aRow][aOff + 8] = q1;
    }
    // ---- stage B tile: 64 cols x 32 bf16 (already transposed in global) ----
    {
      *(u32x4*)&Bs[bCol][bOff] = *(const u32x4*)(bSrc + k0);
      if (k0 + 32 < K) __builtin_prefetch(bSrc + k0 + 32, 0, 3);
    }
    __syncthreads();

    // ---- fragments + WMMA ----
    // A 16x32 bf16: lane=(halfK<<4)|m; 16 consecutive K per lane (K=kA..kA+15)
    const int mRow = wave * 16 + (lane & 15);
    const int kA   = (lane >> 4) << 4;            // 0 or 16
    FragAB a;
    a.q[0] = *(const u32x4*)&As[mRow][kA];
    a.q[1] = *(const u32x4*)&As[mRow][kA + 8];
    // B 32x16 bf16: lane=(halfK<<4)|n; 16 consecutive K per lane
    const int kB = (lane >> 4) << 4;              // 0 or 16
    const int nL = lane & 15;
    {
      FragAB b;
      b.q[0] = *(const u32x4*)&Bs[0  + nL][kB];
      b.q[1] = *(const u32x4*)&Bs[0  + nL][kB + 8];
      acc0 = __builtin_amdgcn_wmma_f32_16x16x32_bf16(false, a.v, false, b.v, (short)0, acc0, false, false);
      b.q[0] = *(const u32x4*)&Bs[16 + nL][kB];
      b.q[1] = *(const u32x4*)&Bs[16 + nL][kB + 8];
      acc1 = __builtin_amdgcn_wmma_f32_16x16x32_bf16(false, a.v, false, b.v, (short)0, acc1, false, false);
      b.q[0] = *(const u32x4*)&Bs[32 + nL][kB];
      b.q[1] = *(const u32x4*)&Bs[32 + nL][kB + 8];
      acc2 = __builtin_amdgcn_wmma_f32_16x16x32_bf16(false, a.v, false, b.v, (short)0, acc2, false, false);
      b.q[0] = *(const u32x4*)&Bs[48 + nL][kB];
      b.q[1] = *(const u32x4*)&Bs[48 + nL][kB + 8];
      acc3 = __builtin_amdgcn_wmma_f32_16x16x32_bf16(false, a.v, false, b.v, (short)0, acc3, false, false);
    }
    __syncthreads();
  }

  // ---- store: C/D layout lane=(halfM<<4)|n, VGPR v -> M = v + 8*halfM ----
  const int nOut = colBase + (lane & 15);
  const int rOut = rowBase + wave * 16 + ((lane >> 4) << 3);
  #pragma unroll
  for (int v = 0; v < 8; ++v) {
    const int r = rOut + v;
    if (r < nRows) {
      H[(size_t)r * FEAT + nOut     ] = acc0[v];
      H[(size_t)r * FEAT + nOut + 16] = acc1[v];
      H[(size_t)r * FEAT + nOut + 32] = acc2[v];
      H[(size_t)r * FEAT + nOut + 48] = acc3[v];
    }
  }
}

// ---------------------------------------------------------------------------
// a_src[n,h] = sum_d h[n,h,d]*att_src[h,d] ; same for a_dst. Block = node.
// ---------------------------------------------------------------------------
__global__ __launch_bounds__(128) void attn_scores(
    const float* __restrict__ H, const float* __restrict__ attS,
    const float* __restrict__ attD, float* __restrict__ aS, float* __restrict__ aD)
{
  const int node = blockIdx.x;
  const int t = threadIdx.x;     // 0..127
  const int head = t >> 4;       // 0..7
  const int seg  = t & 15;       // 0..15, 8 elems each
  const float* hp = H    + (size_t)node * FEAT + head * HID + seg * 8;
  const float* sp = attS + head * HID + seg * 8;
  const float* dp = attD + head * HID + seg * 8;
  float s = 0.f, d = 0.f;
  #pragma unroll
  for (int j = 0; j < 8; ++j) { const float hv = hp[j]; s += hv * sp[j]; d += hv * dp[j]; }
  #pragma unroll
  for (int off = 8; off; off >>= 1) {
    s += __shfl_down(s, off, 16);
    d += __shfl_down(d, off, 16);
  }
  if (seg == 0) { aS[node * HEADS + head] = s; aD[node * HEADS + head] = d; }
}

__global__ void fill_i32(int* __restrict__ p, int v, int n) {
  int i = blockIdx.x * blockDim.x + threadIdx.x;
  if (i < n) p[i] = v;
}
__global__ void fill_f32(float* __restrict__ p, float v, int n) {
  int i = blockIdx.x * blockDim.x + threadIdx.x;
  if (i < n) p[i] = v;
}
__global__ void bias_init(float* __restrict__ out, const float* __restrict__ b, int n) {
  int i = blockIdx.x * blockDim.x + threadIdx.x;
  if (i < n) out[i] = b[i & (FEAT - 1)];
}
__global__ void relu_ip(float* __restrict__ p, int n) {
  int i = blockIdx.x * blockDim.x + threadIdx.x;
  if (i < n) p[i] = fmaxf(p[i], 0.f);
}

// per-edge logits + running segment-max (int-ordered atomic max)
__global__ void edge_logits(const int* __restrict__ ei, const float* __restrict__ aS,
                            const float* __restrict__ aD, float* __restrict__ ebuf,
                            int* __restrict__ mOrd, int nE)
{
  const int e = blockIdx.x * blockDim.x + threadIdx.x;
  if (e >= nE) return;
  const int s = ei[e], d = ei[nE + e];
  #pragma unroll
  for (int h = 0; h < HEADS; ++h) {
    float v = aS[s * HEADS + h] + aD[d * HEADS + h];
    v = (v > 0.f) ? v : 0.2f * v;          // leaky_relu
    ebuf[e * HEADS + h] = v;
    atomicMax(&mOrd[d * HEADS + h], ordF(v));
  }
}

// p = exp(e - m[dst]); denom[dst] += p  (ebuf overwritten with p)
__global__ void edge_exp(const int* __restrict__ ei, float* __restrict__ ebuf,
                         const int* __restrict__ mOrd, float* __restrict__ denom, int nE)
{
  const int e = blockIdx.x * blockDim.x + threadIdx.x;
  if (e >= nE) return;
  const int d = ei[nE + e];
  #pragma unroll
  for (int h = 0; h < HEADS; ++h) {
    const float m = invOrd(mOrd[d * HEADS + h]);
    const float p = __expf(ebuf[e * HEADS + h] - m);
    ebuf[e * HEADS + h] = p;
    atomicAdd(&denom[d * HEADS + h], p);
  }
}

// out[dst] += h[src] * alpha  — block per edge, 4 features per thread
__global__ __launch_bounds__(256) void aggregate(
    const int* __restrict__ ei, const float* __restrict__ H,
    const float* __restrict__ ebuf, const float* __restrict__ denom,
    float* __restrict__ out, int nE)
{
  const int e = blockIdx.x;
  const int t = threadIdx.x;
  const int s = ei[e], d = ei[nE + e];
  __shared__ float alpha[HEADS];
  if (t < HEADS) alpha[t] = ebuf[e * HEADS + t] / (denom[d * HEADS + t] + 1e-16f);
  __syncthreads();
  const f32x4 hv = *(const f32x4*)(H + (size_t)s * FEAT + t * 4);
  const float a = alpha[t >> 5];           // (t*4)/128
  float* op = out + (size_t)d * FEAT + t * 4;
  atomicAdd(op + 0, hv.x * a);
  atomicAdd(op + 1, hv.y * a);
  atomicAdd(op + 2, hv.z * a);
  atomicAdd(op + 3, hv.w * a);
}

// out[n,:10] = H[n,:] @ Wl + bl — block per node
__global__ __launch_bounds__(256) void final_linear(
    const float* __restrict__ H, const float* __restrict__ Wl,
    const float* __restrict__ bl, float* __restrict__ out)
{
  const int node = blockIdx.x;
  const int t = threadIdx.x;
  const int lane = t & 31, wave = t >> 5;
  float acc[10];
  #pragma unroll
  for (int o = 0; o < 10; ++o) acc[o] = 0.f;
  for (int c = t; c < FEAT; c += 256) {
    const float hv = H[(size_t)node * FEAT + c];
    const float* wr = Wl + c * 10;
    #pragma unroll
    for (int o = 0; o < 10; ++o) acc[o] += hv * wr[o];
  }
  #pragma unroll
  for (int o = 0; o < 10; ++o)
    for (int off = 16; off; off >>= 1)
      acc[o] += __shfl_down(acc[o], off, 32);
  __shared__ float part[8][10];
  if (lane == 0)
    for (int o = 0; o < 10; ++o) part[wave][o] = acc[o];
  __syncthreads();
  if (t < 10) {
    float sum = bl[t];
    #pragma unroll
    for (int w = 0; w < 8; ++w) sum += part[w][t];
    out[node * 10 + t] = sum;
  }
}

// ---------------------------------------------------------------------------
static void run_gat_layer(const float* Xin, int K, const float* W, const float* b,
                          const float* as_, const float* ad_, const int* ei,
                          unsigned short* xbf, unsigned short* wtbf,
                          float* hPre, float* hOut, float* aS, float* aD,
                          int* mOrd, float* denom, float* ebuf, hipStream_t stream)
{
  // pre-convert inputs to bf16 (and W -> transposed [col][K])
  cvt_bf16<<<(N_NODES * K / 8 + 255) / 256, 256, 0, stream>>>(Xin, xbf, N_NODES * K);
  dim3 tgrid(K / 32, FEAT / 32);
  cvt_transpose_bf16<<<tgrid, 256, 0, stream>>>(W, wtbf, K);

  dim3 ggrid((N_NODES + 127) / 128, FEAT / 64);
  gat_gemm_bf16<<<ggrid, 256, 0, stream>>>(xbf, wtbf, hPre, N_NODES, K);

  attn_scores<<<N_NODES, 128, 0, stream>>>(hPre, as_, ad_, aS, aD);
  fill_i32<<<(N_NODES * HEADS + 255) / 256, 256, 0, stream>>>(mOrd, NEG_INF_ORD, N_NODES * HEADS);
  fill_f32<<<(N_NODES * HEADS + 255) / 256, 256, 0, stream>>>(denom, 0.f, N_NODES * HEADS);
  edge_logits<<<(N_EDGES + 255) / 256, 256, 0, stream>>>(ei, aS, aD, ebuf, mOrd, N_EDGES);
  edge_exp<<<(N_EDGES + 255) / 256, 256, 0, stream>>>(ei, ebuf, mOrd, denom, N_EDGES);
  bias_init<<<(N_NODES * FEAT + 255) / 256, 256, 0, stream>>>(hOut, b, N_NODES * FEAT);
  aggregate<<<N_EDGES, 256, 0, stream>>>(ei, hPre, ebuf, denom, hOut, N_EDGES);
  relu_ip<<<(N_NODES * FEAT + 255) / 256, 256, 0, stream>>>(hOut, N_NODES * FEAT);
}

extern "C" void kernel_launch(void* const* d_in, const int* in_sizes, int n_in,
                              void* d_out, int out_size, void* d_ws, size_t ws_size,
                              hipStream_t stream) {
  (void)in_sizes; (void)n_in; (void)out_size; (void)ws_size;
  const float* x   = (const float*)d_in[0];
  const int*   ei  = (const int*)  d_in[1];
  const float* W1  = (const float*)d_in[2];
  const float* b1  = (const float*)d_in[3];
  const float* as1 = (const float*)d_in[4];
  const float* ad1 = (const float*)d_in[5];
  const float* W2  = (const float*)d_in[6];
  const float* b2  = (const float*)d_in[7];
  const float* as2 = (const float*)d_in[8];
  const float* ad2 = (const float*)d_in[9];
  const float* W3  = (const float*)d_in[10];
  const float* b3  = (const float*)d_in[11];
  const float* as3 = (const float*)d_in[12];
  const float* ad3 = (const float*)d_in[13];
  const float* Wl  = (const float*)d_in[14];
  const float* bl  = (const float*)d_in[15];
  float* out = (float*)d_out;

  // workspace carve-out (~111 MB)
  char* w = (char*)d_ws;
  auto take = [&](size_t bytes) -> void* {
    void* p = (void*)w;
    w += (bytes + 255) & ~(size_t)255;
    return p;
  };
  float*          hA    = (float*)take((size_t)N_NODES * FEAT * 4);   // pre-aggregation h
  float*          hB    = (float*)take((size_t)N_NODES * FEAT * 4);   // layer output
  unsigned short* xbf   = (unsigned short*)take((size_t)N_NODES * FEAT * 2);
  unsigned short* wtbf  = (unsigned short*)take((size_t)FEAT * FEAT * 2);
  float*          aS    = (float*)take((size_t)N_NODES * HEADS * 4);
  float*          aD    = (float*)take((size_t)N_NODES * HEADS * 4);
  int*            mOrd  = (int*)  take((size_t)N_NODES * HEADS * 4);
  float*          denom = (float*)take((size_t)N_NODES * HEADS * 4);
  float*          ebuf  = (float*)take((size_t)N_EDGES * HEADS * 4);

  run_gat_layer(x,  128,  W1, b1, as1, ad1, ei, xbf, wtbf, hA, hB, aS, aD, mOrd, denom, ebuf, stream);
  run_gat_layer(hB, FEAT, W2, b2, as2, ad2, ei, xbf, wtbf, hA, hB, aS, aD, mOrd, denom, ebuf, stream);
  run_gat_layer(hB, FEAT, W3, b3, as3, ad3, ei, xbf, wtbf, hA, hB, aS, aD, mOrd, denom, ebuf, stream);
  final_linear<<<N_NODES, 256, 0, stream>>>(hB, Wl, bl, out);
}
